// GNN_35167192220464
// MI455X (gfx1250) — matrix operand order repaired and checked by reading, not compiled, not run on hardware
//
#include <hip/hip_runtime.h>
#include <hip/hip_bf16.h>
#include <math.h>

// ---------------------------------------------------------------------------
// Problem constants (match reference)
// ---------------------------------------------------------------------------
#define GH   4            // heads
#define GD   64           // per-head dim
#define GHD  256          // H*D
#define GEMB 64           // head output
#define GFIN 30           // input features
#define GED  11           // edge feature dim
#define GB   1024         // graphs in batch

typedef __attribute__((ext_vector_type(16))) _Float16 v16h;
typedef __attribute__((ext_vector_type(8)))  float    v8f;
typedef __attribute__((ext_vector_type(2)))  __fp16   fp16x2;

__device__ __forceinline__ int imin(int a, int b) { return a < b ? a : b; }

// Monotone float <-> uint encoding so atomicMax(uint) == float max
__device__ __forceinline__ unsigned f2key(float f) {
    unsigned u = __float_as_uint(f);
    return (u & 0x80000000u) ? ~u : (u | 0x80000000u);
}
__device__ __forceinline__ float key2f(unsigned k) {
    unsigned u = (k & 0x80000000u) ? (k ^ 0x80000000u) : ~k;
    return __uint_as_float(u);
}
#define NEGINF_KEY 0x007FFFFFu   // f2key(-inf)

// One v_cvt_pk_rtz_f16_f32 per pair
__device__ __forceinline__ unsigned packh2(float a, float b) {
    union { fp16x2 h; unsigned u; } x;
    x.h = __builtin_amdgcn_cvt_pkrtz(a, b);
    return x.u;
}

// ---------------------------------------------------------------------------
// Generic fill
// ---------------------------------------------------------------------------
__global__ void fill_u32(unsigned* __restrict__ p, unsigned v, long n) {
    long t = (long)blockIdx.x * blockDim.x + threadIdx.x;
    if (t < n) p[t] = v;
}

// ---------------------------------------------------------------------------
// WMMA GEMM: Y[M,N] = X[M,K] @ W[K,N] (+bias) (+=Y) (relu)
// flags: bit0 = add bias, bit1 = accumulate into existing Y, bit2 = relu
// REQUIRES M % 16 == 0 (true for all call sites) -> row guards are
// wave-uniform; K/N tails handled branchlessly (clamped index + select).
//
// Block = 256 threads = 8 waves -> 64x32 tile (4x2 waves of 16x16).
// Hot loop: vector global loads -> cvt_pk f16 -> ds_store -> barrier ->
//           4x ds_load_b128 -> v_wmma_f32_16x16x32_f16.
// ---------------------------------------------------------------------------
#define BM 64
#define BN 32
#define BK 32

__global__ void __launch_bounds__(256)
gemm_wmma(const float* __restrict__ X, const float* __restrict__ W,
          const float* __restrict__ bias, float* __restrict__ Y,
          int M, int K, int Nn, int flags)
{
    __shared__ unsigned sA[BM][BK / 2];   // [row][kpair] : 64x16 u32 (4 KB)
    __shared__ unsigned sB[BN][BK / 2];   // [col][kpair] : 32x16 u32 (2 KB)

    const int tid  = threadIdx.x;
    const int lane = tid & 31;
    const int wave = tid >> 5;          // 0..7
    const int mw   = wave >> 1;         // 0..3 : 16-row sub-tile
    const int nw   = wave & 1;          // 0..1 : 16-col sub-tile
    const int bm   = blockIdx.y * BM;
    const int bn   = blockIdx.x * BN;
    const int half = lane >> 4;         // 0/1
    const int l15  = lane & 15;

    const int rowT = bm + mw * 16;          // wave-uniform tile row base
    const int colC = bn + nw * 16 + l15;    // this lane's C column
    const int colS = imin(colC, Nn - 1);    // clamped (always safe to load)
    const bool cok = colC < Nn;

    // staging coordinates
    const int aRow  = tid >> 2;             // 0..63
    const int aKs   = (tid & 3) * 8;        // 0,8,16,24
    const int gARow = bm + aRow;
    const int sARow = imin(gARow, M - 1);   // clamped row (safe)
    const bool arOK = gARow < M;
    const int bCol  = tid & 31;             // 0..31 (lane==col: coalesced)
    const int bK4   = (tid >> 5) * 4;       // 0,4,...,28
    const int gBCol = bn + bCol;
    const int sBCol = imin(gBCol, Nn - 1);
    const bool bcOK = gBCol < Nn;

    const bool fullM = (bm + BM <= M);

    // ---- C init (wave-uniform row guard; branchless column tail) ----
    v8f c = {};
    if (rowT < M) {
        #pragma unroll
        for (int r = 0; r < 8; ++r) {
            int row = rowT + half * 8 + r;
            float cv = 0.0f;
            if (flags & 2) {
                float y = Y[(size_t)row * Nn + colS];
                cv = cok ? y : 0.0f;
            }
            if (flags & 1) {
                float bb = bias[colS];
                cv += cok ? bb : 0.0f;
            }
            c[r] = cv;
        }
    }

    for (int kk = 0; kk < K; kk += BK) {
        const bool fullK = (kk + BK <= K);

        // ---- stage A (64 rows x 32 k) ----
        if (fullM && fullK) {
            // fast: 2x global_load_b128 + 4x cvt_pk + ds_store_b128
            const float* xp = X + (size_t)gARow * K + kk + aKs;
            float4 x0 = *(const float4*)(xp);
            float4 x1 = *(const float4*)(xp + 4);
            uint4 pk;
            pk.x = packh2(x0.x, x0.y);
            pk.y = packh2(x0.z, x0.w);
            pk.z = packh2(x1.x, x1.y);
            pk.w = packh2(x1.z, x1.w);
            *(uint4*)&sA[aRow][aKs >> 1] = pk;
            if (kk + BK < K) __builtin_prefetch(xp + BK, 0, 3);
        } else {
            // slow: branchless clamped loads + selects (no EXEC branching)
            const float* xr = X + (size_t)sARow * K;
            uint4 pk;
            unsigned pv[4];
            #pragma unroll
            for (int i = 0; i < 4; ++i) {
                int k0 = kk + aKs + 2 * i;
                float x0 = xr[imin(k0,     K - 1)];
                float x1 = xr[imin(k0 + 1, K - 1)];
                x0 = (arOK && k0     < K) ? x0 : 0.0f;
                x1 = (arOK && k0 + 1 < K) ? x1 : 0.0f;
                pv[i] = packh2(x0, x1);
            }
            pk.x = pv[0]; pk.y = pv[1]; pk.z = pv[2]; pk.w = pv[3];
            *(uint4*)&sA[aRow][aKs >> 1] = pk;
        }

        // ---- stage B transposed (32 k x 32 cols -> [col][kpair]) ----
        // single branchless version: 4 coalesced loads + selects
        {
            uint2 pk;
            unsigned pv[2];
            #pragma unroll
            for (int p = 0; p < 2; ++p) {
                int k0 = kk + bK4 + 2 * p;
                float w0 = W[(size_t)imin(k0,     K - 1) * Nn + sBCol];
                float w1 = W[(size_t)imin(k0 + 1, K - 1) * Nn + sBCol];
                w0 = (bcOK && k0     < K) ? w0 : 0.0f;
                w1 = (bcOK && k0 + 1 < K) ? w1 : 0.0f;
                pv[p] = packh2(w0, w1);
            }
            pk.x = pv[0]; pk.y = pv[1];
            *(uint2*)&sB[bCol][bK4 >> 1] = pk;
        }
        __syncthreads();

        // ---- assemble fragments from LDS (4x ds_load_b128 per lane) ----
        // A halves 0..7  = kpairs [half*4 .. half*4+3]
        // A halves 8..15 = kpairs [8+half*4 .. 8+half*4+3]
        // B halves 0..7  = kpairs [half*4 .. half*4+3]
        // B halves 8..15 = kpairs [8+half*4 .. 8+half*4+3]
        const int aFRow = mw * 16 + l15;
        const int bFCol = nw * 16 + l15;
        union { v16h v; uint4 q[2]; } ua, ub;
        ua.q[0] = *(const uint4*)&sA[aFRow][half * 4];
        ua.q[1] = *(const uint4*)&sA[aFRow][8 + half * 4];
        ub.q[0] = *(const uint4*)&sB[bFCol][half * 4];
        ub.q[1] = *(const uint4*)&sB[bFCol][8 + half * 4];

        c = __builtin_amdgcn_wmma_f32_16x16x32_f16(false, ua.v, false, ub.v,
                                                   (short)0, c, false, false);
        __syncthreads();
    }

    // ---- C store: uniform row guard, single divergent column region ----
    if (rowT < M) {
        if (cok) {
            #pragma unroll
            for (int r = 0; r < 8; ++r) {
                int row = rowT + half * 8 + r;
                float v = c[r];
                if (flags & 4) v = fmaxf(v, 0.0f);
                Y[(size_t)row * Nn + colC] = v;
            }
        }
    }
}

// ---------------------------------------------------------------------------
// Edge pass 1: logits[e,h] = (q[dst] . (k[src] + e)) / sqrt(D); segment max
// e computed on the fly: e[hd] = sum_j attr[e,j] * We[j, hd]
// ---------------------------------------------------------------------------
__global__ void edge_logits_max(const float* __restrict__ q,
                                const float* __restrict__ k,
                                const float* __restrict__ attr,
                                const float* __restrict__ We,
                                const int* __restrict__ src,
                                const int* __restrict__ dst,
                                float* __restrict__ logits,
                                unsigned* __restrict__ menc, long EH)
{
    long t = (long)blockIdx.x * blockDim.x + threadIdx.x;
    if (t >= EH) return;
    long e = t >> 2;
    int  h = (int)(t & 3);
    int  s = src[e], d = dst[e];
    const float* qp = q + (size_t)d * GHD + h * GD;
    const float* kp = k + (size_t)s * GHD + h * GD;
    float al[GED];
    #pragma unroll
    for (int j = 0; j < GED; ++j) al[j] = attr[(size_t)e * GED + j];
    float acc = 0.0f;
    for (int dd = 0; dd < GD; ++dd) {
        float ev = 0.0f;
        #pragma unroll
        for (int j = 0; j < GED; ++j) ev += al[j] * We[(size_t)j * GHD + h * GD + dd];
        acc += qp[dd] * (kp[dd] + ev);
    }
    float lg = acc * 0.125f;   // 1/sqrt(64)
    logits[t] = lg;
    atomicMax(&menc[(size_t)d * GH + h], f2key(lg));
}

// ---------------------------------------------------------------------------
// Edge pass 2: ex = exp(logit - m[dst]); denom[dst] += ex (in-place in logits)
// ---------------------------------------------------------------------------
__global__ void edge_exp(const int* __restrict__ dst,
                         const unsigned* __restrict__ menc,
                         float* __restrict__ logits,
                         float* __restrict__ denom, long EH)
{
    long t = (long)blockIdx.x * blockDim.x + threadIdx.x;
    if (t >= EH) return;
    long e = t >> 2;
    int  h = (int)(t & 3);
    int  d = dst[e];
    float m  = key2f(menc[(size_t)d * GH + h]);
    float ex = __expf(logits[t] - m);
    logits[t] = ex;
    atomicAdd(&denom[(size_t)d * GH + h], ex);
}

// ---------------------------------------------------------------------------
// Edge pass 3: out[dst,hd] += alpha * (v[src,hd] + e[hd]); one thread/(edge,hd)
// ---------------------------------------------------------------------------
__global__ void edge_agg(const float* __restrict__ v,
                         const float* __restrict__ attr,
                         const float* __restrict__ We,
                         const int* __restrict__ src,
                         const int* __restrict__ dst,
                         const float* __restrict__ exv,
                         const float* __restrict__ denom,
                         float* __restrict__ outb, long total)
{
    long t = (long)blockIdx.x * blockDim.x + threadIdx.x;
    if (t >= total) return;
    long e  = t >> 8;
    int  hd = (int)(t & 255);
    int  h  = hd >> 6;
    int  s = src[e], d = dst[e];
    float den   = denom[(size_t)d * GH + h];
    float alpha = exv[(size_t)e * GH + h] / fmaxf(den, 1e-16f);
    float ev = 0.0f;
    #pragma unroll
    for (int j = 0; j < GED; ++j)
        ev += attr[(size_t)e * GED + j] * We[(size_t)j * GHD + hd];
    float val = alpha * (v[(size_t)s * GHD + hd] + ev);
    atomicAdd(&outb[(size_t)d * GHD + hd], val);
}

// ---------------------------------------------------------------------------
// Pooling
// ---------------------------------------------------------------------------
__global__ void count_nodes(const int* __restrict__ batch, float* __restrict__ cnt, int N) {
    int n = blockIdx.x * blockDim.x + threadIdx.x;
    if (n < N) atomicAdd(&cnt[batch[n]], 1.0f);
}

__global__ void pool_reduce(const float* __restrict__ h, const int* __restrict__ batch,
                            unsigned* __restrict__ pmax, float* __restrict__ psum, long NF)
{
    long t = (long)blockIdx.x * blockDim.x + threadIdx.x;
    if (t >= NF) return;
    long n = t >> 6;
    int  f = (int)(t & 63);
    int  b = batch[n];
    float v = h[(size_t)n * GEMB + f];
    atomicMax(&pmax[(size_t)b * GEMB + f], f2key(v));
    atomicAdd(&psum[(size_t)b * GEMB + f], v);
}

__global__ void pool_accum(const unsigned* __restrict__ pmax,
                           const float* __restrict__ psum,
                           const float* __restrict__ cnt,
                           float* __restrict__ pooled, long BF)
{
    long t = (long)blockIdx.x * blockDim.x + threadIdx.x;
    if (t >= BF) return;
    long b = t >> 6;
    int  f = (int)(t & 63);
    float mx = key2f(pmax[t]);
    if (!isfinite(mx)) mx = 0.0f;
    pooled[(size_t)b * (2 * GEMB) + f]        += mx;
    pooled[(size_t)b * (2 * GEMB) + GEMB + f] += psum[t] / fmaxf(cnt[b], 1.0f);
}

// ---------------------------------------------------------------------------
// Host-side orchestration
// ---------------------------------------------------------------------------
static inline long cdivl(long a, long b) { return (a + b - 1) / b; }

extern "C" void kernel_launch(void* const* d_in, const int* in_sizes, int n_in,
                              void* d_out, int out_size, void* d_ws, size_t ws_size,
                              hipStream_t stream)
{
    (void)n_in; (void)out_size; (void)ws_size;
    const float* x     = (const float*)d_in[0];
    const float* attr  = (const float*)d_in[1];
    const int*   eidx  = (const int*)d_in[2];
    const int*   batch = (const int*)d_in[3];
    const int N = in_sizes[3];
    const int E = in_sizes[1] / GED;
    const int* src = eidx;
    const int* dst = eidx + E;

    // Param indexing: convs[i] at 4+9i: qw,qb,kw,kb,vw,vb,ew,sw,sb
    // heads[i] at 49+2i: w,b ; lin1: 59,60 ; lin2: 61,62
    auto P = [&](int i) { return (const float*)d_in[i]; };

    // Workspace layout (floats)
    float* ws = (float*)d_ws;
    float*    q      = ws;                       // N*256
    float*    k      = q    + (size_t)N * GHD;   // N*256
    float*    v      = k    + (size_t)N * GHD;   // N*256
    float*    outb   = v    + (size_t)N * GHD;   // N*256
    float*    hA     = outb + (size_t)N * GHD;   // N*64
    float*    hB     = hA   + (size_t)N * GEMB;  // N*64
    float*    att    = hB   + (size_t)N * GEMB;  // E*4 (logits -> exp)
    unsigned* menc   = (unsigned*)(att + (size_t)E * GH);   // N*4
    float*    denom  = (float*)(menc + (size_t)N * GH);     // N*4
    unsigned* pmax   = (unsigned*)(denom + (size_t)N * GH); // B*64
    float*    psum   = (float*)(pmax + (size_t)GB * GEMB);  // B*64
    float*    cnt    = psum + (size_t)GB * GEMB;            // B
    float*    pooled = cnt + GB;                            // B*128
    float*    z1     = pooled + (size_t)GB * 2 * GEMB;      // B*256

    auto gemm = [&](const float* X, const float* W, const float* bias, float* Y,
                    int M, int K, int Nn, int flags) {
        dim3 grid((unsigned)cdivl(Nn, BN), (unsigned)cdivl(M, BM));
        gemm_wmma<<<grid, dim3(256), 0, stream>>>(X, W, bias, Y, M, K, Nn, flags);
    };
    auto fill = [&](void* p, unsigned val, long n) {
        fill_u32<<<dim3((unsigned)cdivl(n, 256)), dim3(256), 0, stream>>>(
            (unsigned*)p, val, n);
    };

    // One-time: pooled accumulator, per-graph node counts
    fill(pooled, 0u, (long)GB * 2 * GEMB);
    fill(cnt, 0u, GB);
    count_nodes<<<dim3((unsigned)cdivl(N, 256)), dim3(256), 0, stream>>>(batch, cnt, N);

    const float* cur = x;
    int din = GFIN;
    float* hping = hA;
    float* hpong = hB;

    for (int L = 0; L < 5; ++L) {
        int cb = 4 + 9 * L;
        const float *qw = P(cb+0), *qb = P(cb+1), *kw = P(cb+2), *kb = P(cb+3);
        const float *vw = P(cb+4), *vb = P(cb+5), *ew = P(cb+6);
        const float *sw = P(cb+7), *sb = P(cb+8);
        const float *hw = P(49 + 2 * L), *hb = P(50 + 2 * L);

        // Dense projections (WMMA)
        gemm(cur, qw, qb, q, N, din, GHD, 1);
        gemm(cur, kw, kb, k, N, din, GHD, 1);
        gemm(cur, vw, vb, v, N, din, GHD, 1);

        // Scatter-softmax attention over edges
        fill(menc, NEGINF_KEY, (long)N * GH);
        fill(denom, 0u, (long)N * GH);
        fill(outb, 0u, (long)N * GHD);
        long EH = (long)E * GH;
        edge_logits_max<<<dim3((unsigned)cdivl(EH, 256)), dim3(256), 0, stream>>>(
            q, k, attr, ew, src, dst, att, menc, EH);
        edge_exp<<<dim3((unsigned)cdivl(EH, 256)), dim3(256), 0, stream>>>(
            dst, menc, att, denom, EH);
        long EHD = (long)E * GHD;
        edge_agg<<<dim3((unsigned)cdivl(EHD, 256)), dim3(256), 0, stream>>>(
            v, attr, ew, src, dst, att, denom, outb, EHD);

        // Skip connection: outb += cur @ sw + sb   (WMMA, accumulate+bias)
        gemm(cur, sw, sb, outb, N, din, GHD, 1 | 2);

        // Head transform: h_next = outb @ hw + hb  (WMMA)
        gemm(outb, hw, hb, hping, N, GHD, GEMB, 1);

        // Pool accumulate
        fill(pmax, NEGINF_KEY, (long)GB * GEMB);
        fill(psum, 0u, (long)GB * GEMB);
        long NF = (long)N * GEMB;
        pool_reduce<<<dim3((unsigned)cdivl(NF, 256)), dim3(256), 0, stream>>>(
            hping, batch, pmax, psum, NF);
        long BF = (long)GB * GEMB;
        pool_accum<<<dim3((unsigned)cdivl(BF, 256)), dim3(256), 0, stream>>>(
            pmax, psum, cnt, pooled, BF);

        cur = hping; din = GEMB;
        float* tmp = hping; hping = hpong; hpong = tmp;
    }

    // Final MLP (WMMA): z1 = relu(pooled @ W1 + b1); out = z1 @ W2 + b2
    gemm(pooled, P(59), P(60), z1, GB, 2 * GEMB, 256, 1 | 4);
    gemm(z1, P(61), P(62), (float*)d_out, GB, 256, 2, 1);
}
